// MultiHeadAttentionLayer_936302870577
// MI455X (gfx1250) — matrix operand order repaired and checked by reading, not compiled, not run on hardware
//
#include <hip/hip_runtime.h>

// ---------------------------------------------------------------------------
// MHA for MI455X (gfx1250): bf16 WMMA path, flash-attention with
// async-to-LDS double-buffered K staging, LDS staging everywhere else.
// ---------------------------------------------------------------------------

#define D_MODEL   1024
#define HEADS     16
#define HEAD_DIM  64
#define SEQ       2048
#define BATCH     4
#define NEG_INF_F (-1000000000.0f)

typedef __attribute__((ext_vector_type(16))) __bf16 v16bf;
typedef __attribute__((ext_vector_type(8)))  __bf16 v8bf;
typedef __attribute__((ext_vector_type(8)))  float  v8f;

__device__ __forceinline__ v8f wmma_bf16(v16bf a, v16bf b, v8f c) {
  return __builtin_amdgcn_wmma_f32_16x16x32_bf16(
      /*neg_a=*/false, a, /*neg_b=*/false, b,
      /*c_mod=*/(short)0, c, /*reuse_a=*/false, /*reuse_b=*/false);
}

// B-fragment (16-bit B 32x16): col = lane&15, element e -> K = e + 16*half.
// 16 contiguous bf16 starting at (row=col, K=16*half) => two b128 loads.
__device__ __forceinline__ v16bf ld_bfrag(const __bf16* p) {
  v16bf r;
  ((v8bf*)&r)[0] = ((const v8bf*)p)[0];
  ((v8bf*)&r)[1] = ((const v8bf*)p)[1];
  return r;
}

// A-fragment (16-bit A 16x32): row = lane&15; VGPR g holds K pair
// 2g + 16*half  => 8 dword loads from the row.
__device__ __forceinline__ v16bf ld_afrag(const __bf16* rowp, int hi) {
  v16bf r;
  unsigned* u = (unsigned*)&r;
#pragma unroll
  for (int g = 0; g < 8; ++g) {
    const int kp = 2 * g + 16 * hi;
    u[g] = *(const unsigned*)(rowp + kp);
  }
  return r;
}

__device__ __forceinline__ v8bf cvt8(float4 a, float4 b) {
  v8bf t;
  t[0] = (__bf16)a.x; t[1] = (__bf16)a.y; t[2] = (__bf16)a.z; t[3] = (__bf16)a.w;
  t[4] = (__bf16)b.x; t[5] = (__bf16)b.y; t[6] = (__bf16)b.z; t[7] = (__bf16)b.w;
  return t;
}

// ---------------------------------------------------------------------------
// Projection GEMM:  Y[M,1024] = X[M,1024] @ W^T + bias,  W row-major [N,K].
// Block 256 threads = 8 waves; tile 64(M) x 128(N); K stepped by 32.
// ---------------------------------------------------------------------------
template <bool BF16_IN, bool PERMUTE_OUT>
__global__ __launch_bounds__(256) void proj_kernel(
    const void* __restrict__ Xv, const float* __restrict__ W,
    const float* __restrict__ bias, void* __restrict__ Yv) {
  __shared__ __align__(16) __bf16 Xs[64][40];
  __shared__ __align__(16) __bf16 Ws[128][40];

  const int tid = threadIdx.x;
  const int wave = tid >> 5;
  const int lane = tid & 31;
  const int lanelo = lane & 15;
  const int hi = lane >> 4;

  const int m0 = blockIdx.x * 64;
  const int n0 = blockIdx.y * 128;

  const int xrow = tid >> 2;        // 0..63
  const int xcol = (tid & 3) * 8;   // 0,8,16,24
  const int wrow = tid >> 1;        // 0..127
  const int wcol = (tid & 1) * 16;  // 0,16

  v8f acc[4] = {};

  for (int k0 = 0; k0 < D_MODEL; k0 += 32) {
    if constexpr (BF16_IN) {
      const __bf16* X = (const __bf16*)Xv;
      *(v8bf*)&Xs[xrow][xcol] =
          *(const v8bf*)(X + (size_t)(m0 + xrow) * D_MODEL + k0 + xcol);
    } else {
      const float* X = (const float*)Xv;
      const float4* src =
          (const float4*)(X + (size_t)(m0 + xrow) * D_MODEL + k0 + xcol);
      *(v8bf*)&Xs[xrow][xcol] = cvt8(src[0], src[1]);
    }
    {
      const float4* src =
          (const float4*)(W + (size_t)(n0 + wrow) * D_MODEL + k0 + wcol);
#pragma unroll
      for (int j = 0; j < 2; ++j) {
        *(v8bf*)&Ws[wrow][wcol + 8 * j] = cvt8(src[2 * j], src[2 * j + 1]);
      }
    }
    __syncthreads();

    const v16bf bfrag = ld_bfrag(&Ws[wave * 16 + lanelo][16 * hi]);
#pragma unroll
    for (int ms = 0; ms < 4; ++ms) {
      const v16bf afrag = ld_afrag(&Xs[ms * 16 + lanelo][0], hi);
      acc[ms] = wmma_bf16(afrag, bfrag, acc[ms]);
    }
    __syncthreads();
  }

  const int n = n0 + wave * 16 + lanelo;
  const float bn = bias[n];
#pragma unroll
  for (int ms = 0; ms < 4; ++ms) {
#pragma unroll
    for (int r = 0; r < 8; ++r) {
      const int m = m0 + ms * 16 + r + 8 * hi;
      const float val = acc[ms][r] + bn;
      if constexpr (PERMUTE_OUT) {
        const int b = m >> 11, s = m & (SEQ - 1);
        const int h = n >> 6, d = n & (HEAD_DIM - 1);
        ((__bf16*)Yv)[(((size_t)(b * HEADS + h)) * SEQ + s) * HEAD_DIM + d] =
            (__bf16)val;
      } else {
        ((float*)Yv)[(size_t)m * D_MODEL + n] = val;
      }
    }
  }
}

// ---------------------------------------------------------------------------
// Flash attention: one block per (b, h, 128 query rows); 8 waves x 16 rows.
// Double-buffered 32-key blocks: K staged with GLOBAL_LOAD_ASYNC_TO_LDS_B128
// (ASYNCcnt), V staged transposed through VGPRs; online softmax.
// ---------------------------------------------------------------------------
__global__ __launch_bounds__(256) void attn_kernel(
    const __bf16* __restrict__ Q, const __bf16* __restrict__ Kmat,
    const __bf16* __restrict__ V, const int* __restrict__ mask,
    __bf16* __restrict__ Aout) {
  __shared__ __align__(16) __bf16 Ks[2][32][72];     // keys x dh, ping-pong
  __shared__ __align__(16) __bf16 Vt[2][64][40];     // dh x keys (transposed)
  __shared__ __align__(16) __bf16 Ps[8][16][40];     // per-wave P scratch

  const int tid = threadIdx.x;
  const int wave = tid >> 5;
  const int lane = tid & 31;
  const int lanelo = lane & 15;
  const int hi = lane >> 4;

  const int b = blockIdx.z;
  const int h = blockIdx.y;
  const int mrow0 = blockIdx.x * 128 + wave * 16;

  const size_t headoff = ((size_t)(b * HEADS + h)) * SEQ * HEAD_DIM;
  const __bf16* qb = Q + headoff;
  const __bf16* kb = Kmat + headoff;
  const __bf16* vb = V + headoff;
  const int* mb = mask + (size_t)b * SEQ * SEQ;

  const v16bf qa0 = ld_afrag(qb + (size_t)(mrow0 + lanelo) * HEAD_DIM + 0, hi);
  const v16bf qa1 = ld_afrag(qb + (size_t)(mrow0 + lanelo) * HEAD_DIM + 32, hi);

  v8f oc[4] = {};
  float mrow[8], lrow[8];
#pragma unroll
  for (int r = 0; r < 8; ++r) { mrow[r] = -1e30f; lrow[r] = 0.0f; }

  const int krow = tid >> 3;       // 0..31
  const int kcol = (tid & 7) * 8;  // 0..56

  // Stage one 32-key block into buffer `buf`.
  auto stage = [&](int kb0, int buf) {
    // K row-major: whole wave's share via one async b128 (32 lanes x 16B).
    {
      const __bf16* gk = kb + (size_t)(kb0 + krow) * HEAD_DIM + kcol;
      const unsigned ldsoff =
          (unsigned)(unsigned long long)(&Ks[buf][krow][kcol]);
      asm volatile("global_load_async_to_lds_b128 %0, %1, off"
                   :: "v"(ldsoff), "v"(gk)
                   : "memory");
    }
    // V transposed (must pass through VGPRs).
    {
      const v8bf tv =
          *(const v8bf*)(vb + (size_t)(kb0 + krow) * HEAD_DIM + kcol);
#pragma unroll
      for (int j = 0; j < 8; ++j) Vt[buf][kcol + j][krow] = tv[j];
    }
    // Prefetch the mask tile for this block (coldest stream).
    __builtin_prefetch(mb + (size_t)(mrow0 + lanelo) * SEQ + kb0 + hi * 16, 0, 3);
  };

  stage(0, 0);

  for (int it = 0; it < SEQ / 32; ++it) {
    const int kb0 = it * 32;
    const int buf = it & 1;

    // Current buffer's async K copy must have landed; then make it (and the
    // VGPR-path V stores) visible to all waves.
    asm volatile("s_wait_asynccnt 0x0" ::: "memory");
    __syncthreads();

    // Kick off next block into the other buffer while we compute.
    if (kb0 + 32 < SEQ) stage(kb0 + 32, buf ^ 1);

    // ---- scores: S(16x32) = Q(16x64) @ K_block^T ----
    v8f sc0, sc1;
    {
      const v8f z = {};
      const v16bf kf0a = ld_bfrag(&Ks[buf][lanelo][0 + 16 * hi]);
      const v16bf kf0b = ld_bfrag(&Ks[buf][lanelo][32 + 16 * hi]);
      sc0 = wmma_bf16(qa0, kf0a, z);
      sc0 = wmma_bf16(qa1, kf0b, sc0);
      const v16bf kf1a = ld_bfrag(&Ks[buf][16 + lanelo][0 + 16 * hi]);
      const v16bf kf1b = ld_bfrag(&Ks[buf][16 + lanelo][32 + 16 * hi]);
      sc1 = wmma_bf16(qa0, kf1a, z);
      sc1 = wmma_bf16(qa1, kf1b, sc1);
    }

    // ---- scale (1/sqrt(64)) + mask ----
#pragma unroll
    for (int r = 0; r < 8; ++r) {
      const int row = mrow0 + r + 8 * hi;
      const size_t mo = (size_t)row * SEQ + kb0;
      float s0 = sc0[r] * 0.125f;
      float s1 = sc1[r] * 0.125f;
      if (mb[mo + lanelo] == 0) s0 = NEG_INF_F;
      if (mb[mo + 16 + lanelo] == 0) s1 = NEG_INF_F;
      sc0[r] = s0;
      sc1[r] = s1;
    }

    // ---- online softmax (row reductions across 16-lane groups) ----
#pragma unroll
    for (int r = 0; r < 8; ++r) {
      float sm = fmaxf(sc0[r], sc1[r]);
#pragma unroll
      for (int off = 1; off < 16; off <<= 1)
        sm = fmaxf(sm, __shfl_xor(sm, off, 32));
      const float mn = fmaxf(mrow[r], sm);
      const float alpha = __expf(mrow[r] - mn);
      mrow[r] = mn;
      const float p0 = __expf(sc0[r] - mn);
      const float p1 = __expf(sc1[r] - mn);
      float rs = p0 + p1;
#pragma unroll
      for (int off = 1; off < 16; off <<= 1) rs += __shfl_xor(rs, off, 32);
      lrow[r] = lrow[r] * alpha + rs;
#pragma unroll
      for (int n = 0; n < 4; ++n) oc[n][r] *= alpha;
      Ps[wave][r + 8 * hi][lanelo] = (__bf16)p0;
      Ps[wave][r + 8 * hi][16 + lanelo] = (__bf16)p1;
    }

    // ---- O += P(16x32) @ V_block(32x64) ----
    const v16bf pa = ld_afrag(&Ps[wave][lanelo][0], hi);
#pragma unroll
    for (int n = 0; n < 4; ++n) {
      const v16bf vf = ld_bfrag(&Vt[buf][n * 16 + lanelo][16 * hi]);
      oc[n] = wmma_bf16(pa, vf, oc[n]);
    }
    // Top-of-loop barrier covers buffer reuse; no trailing barrier needed.
  }

  // ---- normalize and store attention output as bf16 [B,S,D] ----
#pragma unroll
  for (int n = 0; n < 4; ++n) {
#pragma unroll
    for (int r = 0; r < 8; ++r) {
      const int row = mrow0 + r + 8 * hi;
      const int d = n * 16 + lanelo;
      const float val = oc[n][r] / lrow[r];
      Aout[((size_t)b * SEQ + row) * D_MODEL + h * HEAD_DIM + d] = (__bf16)val;
    }
  }
}

// ---------------------------------------------------------------------------
extern "C" void kernel_launch(void* const* d_in, const int* in_sizes, int n_in,
                              void* d_out, int out_size, void* d_ws,
                              size_t ws_size, hipStream_t stream) {
  (void)in_sizes; (void)n_in; (void)out_size; (void)ws_size;

  const float* q  = (const float*)d_in[0];
  const float* k  = (const float*)d_in[1];
  const float* v  = (const float*)d_in[2];
  const int* mask = (const int*)d_in[3];
  const float* Wq = (const float*)d_in[4];
  const float* bq = (const float*)d_in[5];
  const float* Wk = (const float*)d_in[6];
  const float* bk = (const float*)d_in[7];
  const float* Wv = (const float*)d_in[8];
  const float* bv = (const float*)d_in[9];
  const float* Wo = (const float*)d_in[10];
  const float* bo = (const float*)d_in[11];

  const size_t BHSD = (size_t)BATCH * HEADS * SEQ * HEAD_DIM;  // 8,388,608
  __bf16* Qb = (__bf16*)d_ws;
  __bf16* Kb = Qb + BHSD;
  __bf16* Vb = Kb + BHSD;
  __bf16* Ab = Vb + BHSD;  // attention output, bf16 [B,S,D]

  const int M = BATCH * SEQ;            // 8192
  dim3 pblk(256);
  dim3 pgrid(M / 64, D_MODEL / 128);    // 128 x 8

  proj_kernel<false, true><<<pgrid, pblk, 0, stream>>>(q, Wq, bq, Qb);
  proj_kernel<false, true><<<pgrid, pblk, 0, stream>>>(k, Wk, bk, Kb);
  proj_kernel<false, true><<<pgrid, pblk, 0, stream>>>(v, Wv, bv, Vb);

  dim3 agrid(SEQ / 128, HEADS, BATCH);  // 16 x 16 x 4
  attn_kernel<<<agrid, pblk, 0, stream>>>(Qb, Kb, Vb, mask, Ab);

  proj_kernel<true, false><<<pgrid, pblk, 0, stream>>>(Ab, Wo, bo, (float*)d_out);
}
